// TopoAttentionModule_81716047773836
// MI455X (gfx1250) — matrix operand order repaired and checked by reading, not compiled, not run on hardware
//
#include <hip/hip_runtime.h>
#include <hip/hip_bf16.h>
#include <stdint.h>

typedef __attribute__((ext_vector_type(16))) _Float16 v16h;
typedef __attribute__((ext_vector_type(8)))  float    v8f;

#define BB   32
#define CC   32
#define HH   256
#define WW   256
#define PSZ  16
#define HP   16
#define WP   16
#define NN   256
#define HEADS 8
#define DCH   4

// ---------------------------------------------------------------------------
// Kernel 1: patch mean-pool  x(B,C,H,W) -> nf(B,N,C) f32 + f16
// one block per (b, c, hp); b128 streaming loads (64 float4 per row,
// 4 row-subgroups of 64 threads), LDS tree-reduce to 16 patch means
// ---------------------------------------------------------------------------
__global__ void pool_kernel(const float* __restrict__ x,
                            float* __restrict__ nf,
                            _Float16* __restrict__ nf16) {
    int blk = blockIdx.x;
    int hp = blk % HP;
    int c  = (blk / HP) % CC;
    int b  = blk / (HP * CC);
    int t  = threadIdx.x;                       // 256 threads
    int cg = t & 63;                            // float4 column group (0..63)
    int rg = t >> 6;                            // row subgroup (0..3)

    const float4* base = (const float4*)(x + (((size_t)b * CC + c) * HH +
                                              (size_t)hp * PSZ) * WW);
    float4 s = make_float4(0.f, 0.f, 0.f, 0.f);
#pragma unroll
    for (int it = 0; it < 4; ++it) {
        float4 v = base[(size_t)(rg + it * 4) * 64 + cg];
        s.x += v.x; s.y += v.y; s.z += v.z; s.w += v.w;
    }

    __shared__ float4 cs4[4][64];
    cs4[rg][cg] = s;
    __syncthreads();

    if (t < WP) {                               // one thread per patch
        float acc = 0.f;
#pragma unroll
        for (int q = 0; q < 4; ++q) {           // 4 float4 groups per patch
#pragma unroll
            for (int r = 0; r < 4; ++r) {
                float4 v = cs4[r][t * 4 + q];
                acc += v.x + v.y + v.z + v.w;
            }
        }
        acc *= (1.0f / (PSZ * PSZ));
        int node = hp * WP + t;
        size_t idx = ((size_t)b * NN + node) * CC + c;
        nf[idx]   = acc;
        nf16[idx] = (_Float16)acc;
    }
}

// ---------------------------------------------------------------------------
// Kernel 2: pearson correlation -> thresholded adjacency mask (u8)
// one block per batch; xc in f16 LDS; 256 WMMA 16x16 tiles (K=32 exact);
// threshold evaluated as num > 0.5*denom (denom > 0) -- no division
// ---------------------------------------------------------------------------
__global__ void corr_kernel(const float* __restrict__ nf,
                            unsigned char* __restrict__ mask) {
    int b = blockIdx.x;
    int t = threadIdx.x;                        // 256 threads

    __shared__ _Float16 xc16[NN][CC];
    __shared__ float    norms[NN];

    {
        const float* row = nf + ((size_t)b * NN + t) * CC;
        float vals[CC];
        float m = 0.f;
#pragma unroll
        for (int c = 0; c < CC; ++c) { vals[c] = row[c]; m += vals[c]; }
        m *= (1.0f / CC);
        float ss = 0.f;
#pragma unroll
        for (int c = 0; c < CC; ++c) {
            float d = vals[c] - m;
            ss += d * d;
            xc16[t][c] = (_Float16)d;
        }
        norms[t] = sqrtf(ss);
    }
    __syncthreads();

    int wave = t >> 5, lane = t & 31;
    int koff = (lane < 16) ? 0 : 8;             // unused; A also splits K by k0b
    int k0b  = (lane < 16) ? 0 : 16;            // B fragment K split
    int nloc = lane & 15;
    int moff = (lane < 16) ? 0 : 8;

    for (int tile = wave; tile < 256; tile += 8) {
        int ti = tile >> 4, tj = tile & 15;
        int rowA = ti * 16 + (lane & 15);
        int rowB = tj * 16 + (lane & 15);
        v16h a, bf;
#pragma unroll
        for (int u = 0; u < 8; ++u) {
            a[u]     = xc16[rowA][k0b + u];
            a[8 + u] = xc16[rowA][k0b + 8 + u];
        }
#pragma unroll
        for (int u = 0; u < 16; ++u)
            bf[u] = xc16[rowB][k0b + u];        // B[k][n] = xc[j][k]

        v8f acc = {};
        acc = __builtin_amdgcn_wmma_f32_16x16x32_f16(false, a, false, bf,
                                                     (short)0, acc, false, false);
#pragma unroll
        for (int r = 0; r < 8; ++r) {
            int i = ti * 16 + r + moff;
            int j = tj * 16 + nloc;
            float denom = norms[i] * norms[j] + 1e-8f;
            // corr > 0.5  <=>  num > 0.5*denom  (denom > 0)
            mask[((size_t)b * NN + i) * NN + j] =
                (acc[r] > 0.5f * denom || i == j) ? (unsigned char)1
                                                  : (unsigned char)0;
        }
    }
}

// ---------------------------------------------------------------------------
// Kernel 3: linear transforms xl = h@Wl + bl, xr = h@Wr + br (per batch)
// h staged to LDS via GLOBAL_LOAD_ASYNC_TO_LDS_B128 (ASYNCcnt path);
// WMMA 16x16x32 (K = C = 32 exact); 64 tile-tasks over 8 waves
// ---------------------------------------------------------------------------
__global__ void lin_kernel(const _Float16* __restrict__ h16,
                           const float* __restrict__ Wl, const float* __restrict__ bl,
                           const float* __restrict__ Wr, const float* __restrict__ br,
                           float* __restrict__ xl, float* __restrict__ xr,
                           _Float16* __restrict__ xl16) {
    int b = blockIdx.x;
    int t = threadIdx.x;                        // 256 threads

    __shared__ __attribute__((aligned(16))) _Float16 hs[NN][CC];   // 16 KB
    __shared__ _Float16 Wls[CC][CC];
    __shared__ _Float16 Wrs[CC][CC];

    // --- async DMA: copy h16[b] (16 KB) straight into LDS, 16 B per op ---
    {
        uint64_t gbase = (uint64_t)(uintptr_t)(h16 + (size_t)b * NN * CC);
        uint32_t lbase = (uint32_t)(uintptr_t)(&hs[0][0]);   // LDS offset bits
#pragma unroll
        for (int k = 0; k < 4; ++k) {
            uint32_t off = ((uint32_t)t + (uint32_t)k * 256u) * 16u;
            uint64_t ga  = gbase + off;
            uint32_t la  = lbase + off;
            asm volatile("global_load_async_to_lds_b128 %0, %1, off"
                         :: "v"(la), "v"(ga) : "memory");
        }
        asm volatile("s_wait_asynccnt 0x0" ::: "memory");
    }
    for (int k = t; k < CC * CC; k += 256) {
        Wls[k / CC][k % CC] = (_Float16)Wl[k];
        Wrs[k / CC][k % CC] = (_Float16)Wr[k];
    }
    __syncthreads();

    int wave = t >> 5, lane = t & 31;
    int koff = (lane < 16) ? 0 : 8;
    int k0b  = (lane < 16) ? 0 : 16;
    int nloc = lane & 15;
    int moff = (lane < 16) ? 0 : 8;

    for (int tk = wave; tk < 64; tk += 8) {
        int itile = tk >> 2;
        int which = (tk >> 1) & 1;              // 0 = Wl, 1 = Wr
        int ntile = tk & 1;
        int rowA = itile * 16 + (lane & 15);
        int ng   = ntile * 16 + nloc;

        _Float16 (*Wp)[CC] = which ? Wrs : Wls;
        v16h a, bf;
#pragma unroll
        for (int u = 0; u < 8; ++u) {
            a[u]     = hs[rowA][k0b + u];
            a[8 + u] = hs[rowA][k0b + 8 + u];
        }
#pragma unroll
        for (int u = 0; u < 16; ++u)
            bf[u] = Wp[k0b + u][ng];            // B[k][n] = W[cin][cout]

        v8f acc = {};
        acc = __builtin_amdgcn_wmma_f32_16x16x32_f16(false, a, false, bf,
                                                     (short)0, acc, false, false);
        float bias = which ? br[ng] : bl[ng];
#pragma unroll
        for (int r = 0; r < 8; ++r) {
            int i = itile * 16 + r + moff;
            float v = acc[r] + bias;
            size_t idx = ((size_t)b * NN + i) * CC + ng;
            if (which) {
                xr[idx] = v;
            } else {
                xl[idx]   = v;
                xl16[idx] = (_Float16)v;
            }
        }
    }
}

// ---------------------------------------------------------------------------
// Kernel 4: GATv2 attention + aggregation + bias + ELU
// one wave (32 threads) per (b, itile, head); alpha @ xl_h via 8 WMMAs
// ---------------------------------------------------------------------------
__global__ void attn_kernel(const float* __restrict__ xl, const float* __restrict__ xr,
                            const _Float16* __restrict__ xl16,
                            const unsigned char* __restrict__ mask,
                            const float* __restrict__ We, const float* __restrict__ att,
                            const float* __restrict__ bias,
                            float* __restrict__ hout, _Float16* __restrict__ hout16) {
    int blk   = blockIdx.x;                     // b*128 + itile*8 + h
    int h     = blk & 7;
    int itile = (blk >> 3) & 15;
    int b     = blk >> 7;
    int lane  = threadIdx.x;                    // 32 threads = 1 wave
    int i0    = itile * 16;

    __shared__ float    xrh[16][DCH];
    __shared__ float    xlf[NN][DCH];
    __shared__ _Float16 xlh16[NN][DCH];
    __shared__ _Float16 alpha[16][NN];

    for (int idx = lane; idx < 16 * DCH; idx += 32) {
        int i = idx >> 2, d = idx & 3;
        xrh[i][d] = xr[((size_t)b * NN + i0 + i) * CC + h * DCH + d];
    }
    for (int idx = lane; idx < NN * DCH; idx += 32) {
        int j = idx >> 2, d = idx & 3;
        size_t gi = ((size_t)b * NN + j) * CC + h * DCH + d;
        xlf[j][d]   = xl[gi];
        xlh16[j][d] = xl16[gi];
    }
    float ah[DCH], eh[DCH];
#pragma unroll
    for (int d = 0; d < DCH; ++d) { ah[d] = att[h * DCH + d]; eh[d] = We[h * DCH + d]; }
    __syncthreads();

    // masked softmax over j for each of 16 target rows (8 j per lane)
    for (int i = 0; i < 16; ++i) {
        const unsigned char* mrow = mask + ((size_t)b * NN + i0 + i) * NN;
        float l[8];
        float mx = -3.4e38f;
#pragma unroll
        for (int tt = 0; tt < 8; ++tt) {
            int j = lane + 32 * tt;
            float s = 0.f;
#pragma unroll
            for (int d = 0; d < DCH; ++d) {
                float z = xrh[i][d] + xlf[j][d] + eh[d];
                z = (z >= 0.f) ? z : 0.2f * z;          // leaky_relu 0.2
                s += ah[d] * z;
            }
            s = mrow[j] ? s : -1e9f;
            l[tt] = s;
            mx = fmaxf(mx, s);
        }
#pragma unroll
        for (int off = 16; off > 0; off >>= 1) mx = fmaxf(mx, __shfl_xor(mx, off));
        float se = 0.f;
#pragma unroll
        for (int tt = 0; tt < 8; ++tt) { l[tt] = __expf(l[tt] - mx); se += l[tt]; }
#pragma unroll
        for (int off = 16; off > 0; off >>= 1) se += __shfl_xor(se, off);
        float inv = 1.f / se;
#pragma unroll
        for (int tt = 0; tt < 8; ++tt)
            alpha[i][lane + 32 * tt] = (_Float16)(l[tt] * inv);
    }
    __syncthreads();

    // out(16 x 4) = alpha(16 x 256) @ xl_h(256 x 4), 8 K-chunks of 32
    int arow = lane & 15;
    int koff = (lane < 16) ? 0 : 8;
    int k0b  = (lane < 16) ? 0 : 16;
    int nloc = lane & 15;
    v8f acc = {};
#pragma unroll
    for (int kc = 0; kc < 8; ++kc) {
        int jbase = kc * 32;
        v16h a, bf;
#pragma unroll
        for (int u = 0; u < 8; ++u) {
            a[u]     = alpha[arow][jbase + k0b + u];
            a[8 + u] = alpha[arow][jbase + k0b + 8 + u];
        }
#pragma unroll
        for (int u = 0; u < 16; ++u)
            bf[u] = (nloc < DCH) ? xlh16[jbase + k0b + u][nloc] : (_Float16)0.f;
        acc = __builtin_amdgcn_wmma_f32_16x16x32_f16(false, a, false, bf,
                                                     (short)0, acc, false, false);
    }

    if (nloc < DCH) {
        int cg = h * DCH + nloc;
        float bs = bias[cg];
        int moff = (lane < 16) ? 0 : 8;
#pragma unroll
        for (int r = 0; r < 8; ++r) {
            int i = i0 + r + moff;
            float v = acc[r] + bs;
            v = (v > 0.f) ? v : (__expf(v) - 1.f);      // ELU
            size_t idx = ((size_t)b * NN + i) * CC + cg;
            hout[idx]   = v;
            hout16[idx] = (_Float16)v;
        }
    }
}

// ---------------------------------------------------------------------------
// Kernel 5: nearest upsample (x16) + residual add, float4 streaming
// ---------------------------------------------------------------------------
__global__ void upsample_kernel(const float* __restrict__ x,
                                const float* __restrict__ h2,
                                float* __restrict__ out) {
    size_t idx4 = (size_t)blockIdx.x * 256 + threadIdx.x;   // float4 index
    int within  = (int)(idx4 & 63);                         // float4 within row
    size_t r    = idx4 >> 6;                                // b*C*H + c*H + y
    int y = (int)(r & 255);
    int c = (int)((r >> 8) & 31);
    int b = (int)(r >> 13);
    int x0 = within * 4;
    int node = (y >> 4) * WP + (x0 >> 4);
    float add = h2[((size_t)b * NN + node) * CC + c];
    float4 v = ((const float4*)x)[idx4];
    v.x += add; v.y += add; v.z += add; v.w += add;
    ((float4*)out)[idx4] = v;
}

// ---------------------------------------------------------------------------
extern "C" void kernel_launch(void* const* d_in, const int* in_sizes, int n_in,
                              void* d_out, int out_size, void* d_ws, size_t ws_size,
                              hipStream_t stream) {
    const float* x    = (const float*)d_in[0];
    const float* Wl1  = (const float*)d_in[1];
    const float* bl1  = (const float*)d_in[2];
    const float* Wr1  = (const float*)d_in[3];
    const float* br1  = (const float*)d_in[4];
    const float* We1  = (const float*)d_in[5];
    const float* att1 = (const float*)d_in[6];
    const float* b1   = (const float*)d_in[7];
    const float* Wl2  = (const float*)d_in[8];
    const float* bl2  = (const float*)d_in[9];
    const float* Wr2  = (const float*)d_in[10];
    const float* br2  = (const float*)d_in[11];
    const float* We2  = (const float*)d_in[12];
    const float* att2 = (const float*)d_in[13];
    const float* b2   = (const float*)d_in[14];

    char* ws = (char*)d_ws;
    const size_t MB = 1u << 20;
    float*         nf    = (float*)(ws + 0);                    // 1.0 MB
    _Float16*      nf16  = (_Float16*)(ws + 1 * MB);            // 0.5 MB
    unsigned char* msk   = (unsigned char*)(ws + 3 * MB / 2);   // 2.0 MB
    float*         xl    = (float*)(ws + 7 * MB / 2);           // 1.0 MB
    float*         xr    = (float*)(ws + 9 * MB / 2);           // 1.0 MB
    _Float16*      xl16  = (_Float16*)(ws + 11 * MB / 2);       // 0.5 MB
    float*         h1    = (float*)(ws + 6 * MB);               // 1.0 MB
    _Float16*      h116  = (_Float16*)(ws + 7 * MB);            // 0.5 MB
    float*         h2    = (float*)(ws + 15 * MB / 2);          // 1.0 MB
    _Float16*      h216  = (_Float16*)(ws + 17 * MB / 2);       // 0.5 MB

    pool_kernel<<<BB * CC * HP, 256, 0, stream>>>(x, nf, nf16);
    corr_kernel<<<BB, 256, 0, stream>>>(nf, msk);

    lin_kernel<<<BB, 256, 0, stream>>>(nf16, Wl1, bl1, Wr1, br1, xl, xr, xl16);
    attn_kernel<<<BB * 16 * HEADS, 32, 0, stream>>>(xl, xr, xl16, msk,
                                                    We1, att1, b1, h1, h116);

    lin_kernel<<<BB, 256, 0, stream>>>(h116, Wl2, bl2, Wr2, br2, xl, xr, xl16);
    attn_kernel<<<BB * 16 * HEADS, 32, 0, stream>>>(xl, xr, xl16, msk,
                                                    We2, att2, b2, h2, h216);

    upsample_kernel<<<(BB * CC * HH * WW / 4) / 256, 256, 0, stream>>>(x, h2, (float*)d_out);
}
